// EpiNetModel_13142599925761
// MI455X (gfx1250) — matrix operand order repaired and checked by reading, not compiled, not run on hardware
//
#include <hip/hip_runtime.h>
#include <hip/hip_bf16.h>

typedef __bf16 v16bf __attribute__((ext_vector_type(16)));
typedef float  v8f   __attribute__((ext_vector_type(8)));

#define CAP        400000
#define LAT        128
#define BATCH      64
#define ENC_HID    256
#define IN_DIM     784
#define HIDDEN     256
#define NCLASS     100
#define TOPK       8
#define EPS        1e-8f

#define WAVES          4
#define ROWS_PER_WAVE  1024
#define TILES_PER_WAVE 64           // 1024 / 16
#define CHUNK          (WAVES * ROWS_PER_WAVE)   // 4096
#define NBLK           98           // ceil(400000 / 4096)

// ---------------------------------------------------------------------------
// Wave-local LDS synchronization: same-wave LDS ops are in-order on CDNA5
// (ISA 7.3), so store->load through a wave-private LDS region only needs a
// compiler scheduling + memory fence, not a workgroup barrier.
// ---------------------------------------------------------------------------
__device__ __forceinline__ void wave_lds_sync() {
  asm volatile("" ::: "memory");
  __builtin_amdgcn_wave_barrier();
  asm volatile("" ::: "memory");
}

// ---------------------------------------------------------------------------
// top-8 insertion into a descending-sorted register list
// ---------------------------------------------------------------------------
__device__ __forceinline__ void topk_insert(float (&v)[TOPK], int (&ix)[TOPK],
                                            float nv, int ni) {
  if (nv > v[TOPK - 1]) {
    v[TOPK - 1] = nv;
    ix[TOPK - 1] = ni;
#pragma unroll
    for (int q = TOPK - 1; q > 0; --q) {
      if (v[q] > v[q - 1]) {
        float tf = v[q]; v[q] = v[q - 1]; v[q - 1] = tf;
        int   ti = ix[q]; ix[q] = ix[q - 1]; ix[q - 1] = ti;
      }
    }
  }
}

// ---------------------------------------------------------------------------
// Kernel 1: encoder  x[64,784] -> z[64,128], zn = l2norm(z) in bf16
// ---------------------------------------------------------------------------
__global__ void __launch_bounds__(256)
enc_kernel(const float* __restrict__ x,
           const float* __restrict__ w1, const float* __restrict__ b1,
           const float* __restrict__ w2, const float* __restrict__ b2,
           float* __restrict__ zout, __bf16* __restrict__ znout) {
  __shared__ float xs[IN_DIM];
  __shared__ float hs[ENC_HID];
  __shared__ float zs[LAT];
  __shared__ float zinv;
  const int b = blockIdx.x, tid = threadIdx.x;

  for (int i = tid; i < IN_DIM; i += 256) xs[i] = x[b * IN_DIM + i];
  __syncthreads();

  float acc = b1[tid];
  for (int k = 0; k < IN_DIM; ++k) acc = fmaf(xs[k], w1[k * ENC_HID + tid], acc);
  hs[tid] = fmaxf(acc, 0.0f);
  __syncthreads();

  if (tid < LAT) {
    float a2 = b2[tid];
    for (int k = 0; k < ENC_HID; ++k) a2 = fmaf(hs[k], w2[k * LAT + tid], a2);
    zs[tid] = a2;
    zout[b * LAT + tid] = a2;
  }
  __syncthreads();

  if (tid == 0) {
    float ss = 0.0f;
    for (int k = 0; k < LAT; ++k) ss += zs[k] * zs[k];
    zinv = rsqrtf(ss + EPS);
  }
  __syncthreads();

  if (tid < LAT) znout[b * LAT + tid] = (__bf16)(zs[tid] * zinv);
}

// ---------------------------------------------------------------------------
// Kernel 2: scorer — streaming bf16 WMMA over c_buffer + per-block top-8
//
// Data path: global b128 -> VGPR -> v_cvt_pk_bf16_f32 -> v_wmma (no LDS
// staging; a 16x128 tile is covered exactly once by the 32 lanes x 4 k-chunks
// of the B layout). LDS is used only for the 64x16 score transpose (top-k),
// which is wave-private -> wave-local sync, no block barriers in the stream.
// ---------------------------------------------------------------------------
struct Tile { float4 v[16]; };     // lane-private half-row: 4 k-chunks x 16 f32

__device__ __forceinline__ Tile load_tile(const float* __restrict__ cbuf,
                                          long long rbase, int l16, int lhalf) {
  Tile t;
  const long long row = rbase + l16;
  const bool valid = row < CAP;
  const float4 z4 = {0.f, 0.f, 0.f, 0.f};
  const float4* src =
      reinterpret_cast<const float4*>(cbuf + row * LAT + lhalf * 16);
#pragma unroll
  for (int kc = 0; kc < 4; ++kc) {
#pragma unroll
    for (int j = 0; j < 4; ++j) {
      // column (floats) = kc*32 + j*4 relative to lane base
      t.v[kc * 4 + j] = valid ? src[kc * 8 + j] : z4;
    }
  }
  return t;
}

struct ScorerSmem {
  union U {
    float sc[WAVES][BATCH][17];                  // wave-private score transpose
    struct {
      float mv[WAVES][BATCH][TOPK];              // merge-phase overlays
      int   mi[WAVES][BATCH][TOPK];
    } mg;
  } u;
};

__global__ void __launch_bounds__(WAVES * 32)
scorer_kernel(const float* __restrict__ cbuf,
              const float* __restrict__ r0,
              const float* __restrict__ tau,
              const __bf16* __restrict__ zn,
              float* __restrict__ pvals, int* __restrict__ pidx) {
  __shared__ ScorerSmem sm;

  const int tid   = threadIdx.x;
  const int w     = tid >> 5;
  const int lane  = tid & 31;
  const int lhalf = lane >> 4;   // 0 | 1
  const int l16   = lane & 15;
  const long long waveBase =
      (long long)blockIdx.x * CHUNK + (long long)w * ROWS_PER_WAVE;

  // ---- loop-invariant A fragments: zn is the 64x128 bf16 A operand ----
  // A k order used here (contiguous, B matches): lane L row M = L%16;
  // k(p,s) = kc*32 + (L>=16)*16 + p*2 + s
  v16bf afrag[4][4];
#pragma unroll
  for (int mt = 0; mt < 4; ++mt) {
#pragma unroll
    for (int kc = 0; kc < 4; ++kc) {
      v16bf a;
#pragma unroll
      for (int p = 0; p < 8; ++p) {
#pragma unroll
        for (int s = 0; s < 2; ++s) {
          const int k = kc * 32 + lhalf * 16 + p * 2 + s;
          a[2 * p + s] = zn[(mt * 16 + l16) * LAT + k];
        }
      }
      afrag[mt][kc] = a;
    }
  }

  // per-lane top-8 for batch rows 2*lane and 2*lane+1
  float tv[2][TOPK];
  int   ti[2][TOPK];
#pragma unroll
  for (int r = 0; r < 2; ++r)
#pragma unroll
    for (int q = 0; q < TOPK; ++q) { tv[r][q] = -3.0e38f; ti[r][q] = 0; }

  // double-buffered tile stream
  Tile cur = load_tile(cbuf, waveBase, l16, lhalf);

  for (int t = 0; t < TILES_PER_WAVE; ++t) {
    const long long rbase = waveBase + (long long)t * 16;

    // issue next tile's 16 global b128 loads before touching `cur`
    Tile nxt = cur;
    if (t + 1 < TILES_PER_WAVE)
      nxt = load_tile(cbuf, rbase + 16, l16, lhalf);

    // L2 warm-up two tiles ahead (global_prefetch_b8); RT policy on purpose:
    // c_buffer (204.8 MB) nearly fits the 192 MB L2 across graph replays.
    if (t + 2 < TILES_PER_WAVE) {
      const long long pb = rbase + 32;
      if (pb + l16 < CAP)
        __builtin_prefetch(cbuf + (pb + l16) * LAT + lhalf * 16, 0, 1);
    }

    // ---- row norm from registers; combine half-rows via shfl_xor(16) ----
    float ss = 0.0f;
#pragma unroll
    for (int i = 0; i < 16; ++i) {
      const float4 q = cur.v[i];
      ss = fmaf(q.x, q.x, ss);
      ss = fmaf(q.y, q.y, ss);
      ss = fmaf(q.z, q.z, ss);
      ss = fmaf(q.w, q.w, ss);
    }
    ss += __shfl_xor(ss, 16, 32);              // lane n <-> lane n+16

    const long long row = rbase + l16;
    const bool nvalid = row < CAP;
    float sal = 0.0f;
    if (nvalid) sal = r0[row] * __expf(-0.01f * tau[row]);
    const float scl = sal * rsqrtf(ss + EPS);  // lane owns output column n=l16

    // ---- WMMA: 4 M-tiles x 4 K-chunks, bf16 in, f32 accumulate ----
    v8f acc[4];
#pragma unroll
    for (int mt = 0; mt < 4; ++mt)
      acc[mt] = (v8f){0.f, 0.f, 0.f, 0.f, 0.f, 0.f, 0.f, 0.f};

#pragma unroll
    for (int kc = 0; kc < 4; ++kc) {
      // B k order used here (contiguous, matches A): lane L col N = L%16; k = kc*32 + (L>=16)*16 + e
      v16bf bfrag;
#pragma unroll
      for (int j = 0; j < 4; ++j) {
        const float4 q = cur.v[kc * 4 + j];
        bfrag[4 * j + 0] = (__bf16)q.x;
        bfrag[4 * j + 1] = (__bf16)q.y;
        bfrag[4 * j + 2] = (__bf16)q.z;
        bfrag[4 * j + 3] = (__bf16)q.w;
      }
#pragma unroll
      for (int mt = 0; mt < 4; ++mt) {
        acc[mt] = __builtin_amdgcn_wmma_f32_16x16x32_bf16(
            false, afrag[mt][kc], false, bfrag, (short)0, acc[mt], false, false);
      }
    }

    // ---- scale columns, transpose through wave-private LDS ----
#pragma unroll
    for (int mt = 0; mt < 4; ++mt) {
#pragma unroll
      for (int v = 0; v < 8; ++v) {
        const int m = mt * 16 + v + 8 * lhalf;   // C layout: M = v + 8*(L/16)
        sm.u.sc[w][m][l16] = nvalid ? acc[mt][v] * scl : -3.0e38f;
      }
    }
    wave_lds_sync();   // same-wave LDS ops are in-order; no block barrier

    // ---- per-lane top-8 update over this tile's 16 columns ----
#pragma unroll
    for (int r2 = 0; r2 < 2; ++r2) {
      const int m = 2 * lane + r2;
#pragma unroll
      for (int n = 0; n < 16; ++n) {
        topk_insert(tv[r2], ti[r2], sm.u.sc[w][m][n], (int)(rbase + n));
      }
    }
    wave_lds_sync();   // reads complete (in-order) before next tile rewrites sc

    cur = nxt;
  }

  // All waves must finish streaming before the merge overlay (union over all
  // waves' sc regions) is written — this is the only cross-wave sync point.
  __syncthreads();

  // ---- block merge: 4 wave lists -> one top-8 per batch row ----
#pragma unroll
  for (int r2 = 0; r2 < 2; ++r2) {
    const int m = 2 * lane + r2;
#pragma unroll
    for (int q = 0; q < TOPK; ++q) {
      sm.u.mg.mv[w][m][q] = tv[r2][q];
      sm.u.mg.mi[w][m][q] = ti[r2][q];
    }
  }
  __syncthreads();

  if (tid < BATCH) {
    const int m = tid;
    float bv[TOPK]; int bi[TOPK];
#pragma unroll
    for (int q = 0; q < TOPK; ++q) { bv[q] = -3.0e38f; bi[q] = 0; }
    for (int ww = 0; ww < WAVES; ++ww)
      for (int q = 0; q < TOPK; ++q)
        topk_insert(bv, bi, sm.u.mg.mv[ww][m][q], sm.u.mg.mi[ww][m][q]);
#pragma unroll
    for (int q = 0; q < TOPK; ++q) {
      pvals[((long long)blockIdx.x * BATCH + m) * TOPK + q] = bv[q];
      pidx [((long long)blockIdx.x * BATCH + m) * TOPK + q] = bi[q];
    }
  }
}

// ---------------------------------------------------------------------------
// Kernel 3: global top-8 merge + recall gather -> zr[64,256]
// ---------------------------------------------------------------------------
__global__ void __launch_bounds__(64)
merge_kernel(const float* __restrict__ pvals, const int* __restrict__ pidx,
             const float* __restrict__ zbuf, const float* __restrict__ z,
             float* __restrict__ zr, int nblk) {
  const int m = threadIdx.x;   // batch row
  float bv[TOPK]; int bi[TOPK];
#pragma unroll
  for (int q = 0; q < TOPK; ++q) { bv[q] = -3.0e38f; bi[q] = 0; }

  for (int bk = 0; bk < nblk; ++bk) {
    const long long base = ((long long)bk * BATCH + m) * TOPK;
#pragma unroll
    for (int q = 0; q < TOPK; ++q)
      topk_insert(bv, bi, pvals[base + q], pidx[base + q]);
  }

  float sumw = 0.0f;
#pragma unroll
  for (int q = 0; q < TOPK; ++q) sumw += bv[q];
  const float inv = 1.0f / (sumw + EPS);

  for (int d = 0; d < LAT; ++d) {
    float r = 0.0f;
#pragma unroll
    for (int q = 0; q < TOPK; ++q)
      r = fmaf(bv[q], zbuf[(long long)bi[q] * LAT + d], r);
    zr[m * 2 * LAT + LAT + d] = r * inv;
    zr[m * 2 * LAT + d]       = z[m * LAT + d];
  }
}

// ---------------------------------------------------------------------------
// Kernel 4: decoder  [z;r] -> relu MLP -> logits[64,100]
// ---------------------------------------------------------------------------
__global__ void __launch_bounds__(256)
dec_kernel(const float* __restrict__ zr,
           const float* __restrict__ w1, const float* __restrict__ b1,
           const float* __restrict__ w2, const float* __restrict__ b2,
           float* __restrict__ out) {
  __shared__ float zrs[2 * LAT];
  __shared__ float hs[HIDDEN];
  const int m = blockIdx.x, tid = threadIdx.x;

  zrs[tid] = zr[m * 2 * LAT + tid];
  __syncthreads();

  float acc = b1[tid];
  for (int k = 0; k < 2 * LAT; ++k) acc = fmaf(zrs[k], w1[k * HIDDEN + tid], acc);
  hs[tid] = fmaxf(acc, 0.0f);
  __syncthreads();

  if (tid < NCLASS) {
    float l = b2[tid];
    for (int k = 0; k < HIDDEN; ++k) l = fmaf(hs[k], w2[k * NCLASS + tid], l);
    out[m * NCLASS + tid] = l;
  }
}

// ---------------------------------------------------------------------------
// Launch
// ---------------------------------------------------------------------------
extern "C" void kernel_launch(void* const* d_in, const int* in_sizes, int n_in,
                              void* d_out, int out_size, void* d_ws, size_t ws_size,
                              hipStream_t stream) {
  (void)in_sizes; (void)n_in; (void)out_size; (void)ws_size;

  const float* x    = (const float*)d_in[0];
  const float* cbuf = (const float*)d_in[1];
  const float* zbuf = (const float*)d_in[2];
  const float* r0   = (const float*)d_in[3];
  const float* tau  = (const float*)d_in[4];
  const float* ew1  = (const float*)d_in[5];
  const float* eb1  = (const float*)d_in[6];
  const float* ew2  = (const float*)d_in[7];
  const float* eb2  = (const float*)d_in[8];
  const float* dw1  = (const float*)d_in[9];
  const float* db1  = (const float*)d_in[10];
  const float* dw2  = (const float*)d_in[11];
  const float* db2  = (const float*)d_in[12];

  char* ws = (char*)d_ws;
  float*  z     = (float*) (ws + 0);                 // 64*128 f32   = 32768 B
  __bf16* zn    = (__bf16*)(ws + 32768);             // 64*128 bf16  = 16384 B
  float*  pvals = (float*) (ws + 49152);             // 98*64*8 f32  = 200704 B
  int*    pidx  = (int*)   (ws + 249856);            // 98*64*8 i32  = 200704 B
  float*  zr    = (float*) (ws + 450560);            // 64*256 f32   = 65536 B

  enc_kernel<<<BATCH, 256, 0, stream>>>(x, ew1, eb1, ew2, eb2, z, zn);
  scorer_kernel<<<NBLK, WAVES * 32, 0, stream>>>(cbuf, r0, tau, zn, pvals, pidx);
  merge_kernel<<<1, BATCH, 0, stream>>>(pvals, pidx, zbuf, z, zr, NBLK);
  dec_kernel<<<BATCH, 256, 0, stream>>>(zr, dw1, db1, dw2, db2, (float*)d_out);
}
